// Block_25572235281069
// MI455X (gfx1250) — matrix operand order repaired and checked by reading, not compiled, not run on hardware
//
#include <hip/hip_runtime.h>

typedef __attribute__((ext_vector_type(16))) _Float16 v16h;
typedef __attribute__((ext_vector_type(8)))  float    v8f;

#define WAVES_PER_WG   4
#define WG_THREADS     (WAVES_PER_WG * 32)
#define LDS_F_PER_WAVE 2560   // 10 KB per wave

#define WMMA_F16(a, b, c) \
  __builtin_amdgcn_wmma_f32_16x16x32_f16(false, (a), false, (b), (short)0, (c), false, false)

// LDS ops are in-order within a wave; compiler fence + hw drain so cross-lane
// LDS producer->consumer handoffs inside one wave are safe.
__device__ __forceinline__ void wave_sync() {
  asm volatile("s_wait_dscnt 0x0" ::: "memory");
}

// A-matrix (16x32 f16) gather, contiguous K order (matches gather_b):
// lanes 0-15: row=ln, h[0..15]=K 0..15
// lanes 16-31: row=ln, h[0..15]=K 16..31
__device__ __forceinline__ v16h gather_a(const float* p, int stride, int c0,
                                         int ln, int sub) {
  const float* row = p + ln * stride + c0 + sub * 16;
  v16h a;
#pragma unroll
  for (int j = 0; j < 8; ++j) {
    a[j]     = (_Float16)row[j];
    a[8 + j] = (_Float16)row[8 + j];
  }
  return a;
}

// B-matrix (32x16 f16) gather: lanes 0-15 hold K=0..15 of col=ln,
// lanes 16-31 hold K=16..31.
__device__ __forceinline__ v16h gather_b(const float* w, int stride, int k0,
                                         int col, int sub) {
  const float* p = w + (size_t)(k0 + sub * 16) * stride + col;
  v16h b;
#pragma unroll
  for (int j = 0; j < 16; ++j) b[j] = (_Float16)p[j * stride];
  return b;
}

__global__ __launch_bounds__(WG_THREADS) void
transformer_block_kernel(const float* __restrict__ X,
                         const float* __restrict__ Wattn,   // (4,32,24)
                         const float* __restrict__ Wproj,   // (32,32)
                         const float* __restrict__ Wff1,    // (32,128)
                         const float* __restrict__ Wff2,    // (128,32)
                         float* __restrict__ Out, int ntiles) {
  // per-wave working tiles: 40 KB
  __shared__ __align__(16) float smem[WAVES_PER_WG * LDS_F_PER_WAVE];
  // FF weights as f16 B-layout register images, shared by all waves: 16 KB
  // tiles 0..7  = Bff1[nt], tiles 8..15 = Bff2[kt*2+nt]; 512 halves/tile
  __shared__ __align__(16) _Float16 wsmem[16 * 512];

  const int tid  = threadIdx.x;
  const int wave = tid >> 5;
  const int lane = tid & 31;
  const int sub  = lane >> 4;   // which 16-lane half
  const int ln   = lane & 15;

  float* Xs  = &smem[wave * LDS_F_PER_WAVE]; // 512  f32: 16x32 X tile (2 batches)
  float* QKV = Xs + 512;                     // 1536 f32: 16x96
  float* ATT = Xs + 2048;                    // 512  f32: 16x32
  float* Hs  = Xs + 512;                     // 2048 f32: 16x128 (overlays QKV+ATT)

  // ---- stage FF weights into LDS as ready-to-load B-register images ----
  // wave w converts tiles 4w..4w+3 (each lane emits its own 32-byte chunk)
#pragma unroll
  for (int i = 0; i < 4; ++i) {
    const int t = wave * 4 + i;
    v16h b;
    if (t < 8) {                     // Bff1 tile t: cols t*16.., K=0..31
      b = gather_b(Wff1, 128, 0, t * 16 + ln, sub);
    } else {                         // Bff2 tile: kt=(t-8)/2, nt=(t-8)&1
      const int kt = (t - 8) >> 1, nt = (t - 8) & 1;
      b = gather_b(Wff2, 32, kt * 32, nt * 16 + ln, sub);
    }
    *(v16h*)(wsmem + t * 512 + lane * 16) = b;
  }
  __syncthreads();   // one-time; loop below has no workgroup barriers

  // ---- latency-critical attention weights stay register-resident ----
  v16h Bqkv[6];
#pragma unroll
  for (int t = 0; t < 6; ++t) {
    const int g  = t * 16 + ln;      // fused col over heads: g = head*24 + d
    const int hh = g / 24, d = g % 24;
    Bqkv[t] = gather_b(Wattn + hh * 768, 24, 0, d, sub);
  }
  v16h Bproj[2];
#pragma unroll
  for (int t = 0; t < 2; ++t) Bproj[t] = gather_b(Wproj, 32, 0, t * 16 + ln, sub);

  const _Float16* wbase = wsmem + lane * 16;   // this lane's chunk of each tile
  const unsigned ldsXs = (unsigned)(unsigned long long)(void*)Xs;
  const int waveGlobal = blockIdx.x * WAVES_PER_WG + wave;
  const int waveCount  = gridDim.x * WAVES_PER_WG;

  for (int tile = waveGlobal; tile < ntiles; tile += waveCount) {
    const float* xg = X + (size_t)tile * 512;
    float*       og = Out + (size_t)tile * 512;

    // previous iteration's async store must finish reading Xs first
    asm volatile("s_wait_asynccnt 0x0" ::: "memory");

    // ---- async VGPR-bypassing copy: 2 KB X tile -> LDS (coalesced 512B/op) ----
#pragma unroll
    for (int j = 0; j < 4; ++j) {
      unsigned loff = ldsXs + (unsigned)((j * 32 + lane) * 16);
      const float* ga = xg + (j * 32 + lane) * 4;
      asm volatile("global_load_async_to_lds_b128 %0, %1, off"
                   :: "v"(loff), "v"(ga) : "memory");
    }
    asm volatile("s_wait_asynccnt 0x0" ::: "memory");

    // ---- fused per-head QKV: (16x32)x(32x96) = 6 WMMAs, K=32 exact ----
    v16h Ax = gather_a(Xs, 32, 0, ln, sub);
#pragma unroll
    for (int t = 0; t < 6; ++t) {
      v8f c = {};
      c = WMMA_F16(Ax, Bqkv[t], c);
#pragma unroll
      for (int r = 0; r < 8; ++r)
        QKV[(sub * 8 + r) * 96 + t * 16 + ln] = c[r];
    }
    wave_sync();

    // ---- causal attention: 64 (batch,head,qrow) tasks, 2 per lane ----
#pragma unroll
    for (int task = 0; task < 2; ++task) {
      const int id = lane + task * 32;
      const int bi = id >> 5;          // batch-in-tile
      const int hh = (id >> 3) & 3;    // head
      const int tq = id & 7;           // query row
      const float* hb = QKV + (bi * 8) * 96 + hh * 24;  // cols: K|Q|V (8 each)
      const float4 q0 = *(const float4*)(hb + tq * 96 + 8);
      const float4 q1 = *(const float4*)(hb + tq * 96 + 12);
      float sc[8];
      float m = -3.0e38f;
#pragma unroll
      for (int j = 0; j < 8; ++j) {
        const float4 k0 = *(const float4*)(hb + j * 96);
        const float4 k1 = *(const float4*)(hb + j * 96 + 4);
        float s = q0.x*k0.x + q0.y*k0.y + q0.z*k0.z + q0.w*k0.w +
                  q1.x*k1.x + q1.y*k1.y + q1.z*k1.z + q1.w*k1.w;
        s *= 0.17677669529663687f;     // c^-0.5 = 1/sqrt(32), faithful to ref
        s = (j <= tq) ? s : -3.0e38f;  // causal mask
        sc[j] = s;
        m = fmaxf(m, s);
      }
      float sum = 0.f;
#pragma unroll
      for (int j = 0; j < 8; ++j) {
        float e = __expf(sc[j] - m);
        e = (j <= tq) ? e : 0.f;
        sc[j] = e;
        sum += e;
      }
      const float inv = 1.0f / sum;
      float a0x=0,a0y=0,a0z=0,a0w=0,a1x=0,a1y=0,a1z=0,a1w=0;
#pragma unroll
      for (int j = 0; j < 8; ++j) {
        const float w = sc[j] * inv;
        const float4 v0 = *(const float4*)(hb + j * 96 + 16);
        const float4 v1 = *(const float4*)(hb + j * 96 + 20);
        a0x += w * v0.x; a0y += w * v0.y; a0z += w * v0.z; a0w += w * v0.w;
        a1x += w * v1.x; a1y += w * v1.y; a1z += w * v1.z; a1w += w * v1.w;
      }
      float* op = ATT + (bi * 8 + tq) * 32 + hh * 8;   // concat heads
      *(float4*)(op)     = make_float4(a0x, a0y, a0z, a0w);
      *(float4*)(op + 4) = make_float4(a1x, a1y, a1z, a1w);
    }
    wave_sync();

    // ---- out-projection (2 WMMAs) + residual 1 (f32, in-place in Xs) ----
    v16h Aatt = gather_a(ATT, 32, 0, ln, sub);
    v8f p0 = {}, p1 = {};
    p0 = WMMA_F16(Aatt, Bproj[0], p0);
    p1 = WMMA_F16(Aatt, Bproj[1], p1);
#pragma unroll
    for (int r = 0; r < 8; ++r) {
      const int row = (sub * 8 + r) * 32;
      Xs[row + ln]      += p0[r];
      Xs[row + 16 + ln] += p1[r];
    }
    wave_sync();

    // ---- FF1 + ReLU: (16x32)x(32x128) = 8 WMMAs; B tiles reloaded from LDS ----
    v16h Ax1 = gather_a(Xs, 32, 0, ln, sub);
#pragma unroll
    for (int nt = 0; nt < 8; ++nt) {
      const v16h bf = *(const v16h*)(wbase + nt * 512);   // 2x ds_load_b128
      v8f c = {};
      c = WMMA_F16(Ax1, bf, c);
#pragma unroll
      for (int r = 0; r < 8; ++r)
        Hs[(sub * 8 + r) * 128 + nt * 16 + ln] = fmaxf(c[r], 0.f);
    }
    wave_sync();

    // ---- FF2: (16x128)x(128x32), K chained over 4 WMMAs/acc, + residual 2 ----
    v8f d0 = {}, d1 = {};
#pragma unroll
    for (int kt = 0; kt < 4; ++kt) {
      v16h Ah = gather_a(Hs, 128, kt * 32, ln, sub);
      const v16h b0 = *(const v16h*)(wbase + (8 + kt * 2 + 0) * 512);
      const v16h b1 = *(const v16h*)(wbase + (8 + kt * 2 + 1) * 512);
      d0 = WMMA_F16(Ah, b0, d0);
      d1 = WMMA_F16(Ah, b1, d1);
    }
#pragma unroll
    for (int r = 0; r < 8; ++r) {
      const int row = (sub * 8 + r) * 32;
      Xs[row + ln]      += d0[r];
      Xs[row + 16 + ln] += d1[r];
    }
    wave_sync();

    // ---- async VGPR-bypassing store: LDS -> global (coalesced) ----
#pragma unroll
    for (int j = 0; j < 4; ++j) {
      unsigned loff = ldsXs + (unsigned)((j * 32 + lane) * 16);
      float* ga = og + (j * 32 + lane) * 4;
      asm volatile("global_store_async_from_lds_b128 %0, %1, off"
                   :: "v"(ga), "v"(loff) : "memory");
    }
  }
}

extern "C" void kernel_launch(void* const* d_in, const int* in_sizes, int n_in,
                              void* d_out, int out_size, void* d_ws, size_t ws_size,
                              hipStream_t stream) {
  const float* X     = (const float*)d_in[0];
  const float* Wattn = (const float*)d_in[1];
  const float* Wproj = (const float*)d_in[2];
  const float* Wff1  = (const float*)d_in[3];
  const float* Wff2  = (const float*)d_in[4];
  float* out = (float*)d_out;

  const int b      = in_sizes[0] / 256;  // X is (b, 8, 32)
  const int ntiles = b / 2;              // 2 batches per 16-row WMMA tile
  int grid = 1024;                       // 4096 waves -> ~32 tiles each at b=262144
  if (grid * WAVES_PER_WG > ntiles)
    grid = (ntiles + WAVES_PER_WG - 1) / WAVES_PER_WG;
  if (grid < 1) grid = 1;

  transformer_block_kernel<<<grid, WG_THREADS, 0, stream>>>(
      X, Wattn, Wproj, Wff1, Wff2, out, ntiles);
}
